// MyParticleNetwork_40020505264371
// MI455X (gfx1250) — matrix operand order repaired and checked by reading, not compiled, not run on hardware
//
#include <hip/hip_runtime.h>

// ---------------------------------------------------------------------------
// Continuous convolution (Open3D CConv) for MI455X / gfx1250, wave32.
// Phase 1: per-lane particle, gather neighbors, poly6 window + ball->cube +
//          trilinear scatter into per-lane LDS accumulator A[192] (f32).
// Phase 2: per-wave WMMA contraction  out[16x16] = A[16x192] @ W[192x16pad]
//          using 6x v_wmma_f32_16x16x32_f16 per 16-particle tile.
// ---------------------------------------------------------------------------

#define N_PART   50000
#define KMAX     80
#define RADIUS_F 0.3f
#define KDIM     192          // 64 taps * 3 input channels
#define ASTRIDE  193          // padded LDS row stride (conflict-free: 193 % 64 == 1)
#define BLOCK    64           // 2 waves of 32

typedef __attribute__((ext_vector_type(16))) _Float16 v16h;
typedef __attribute__((ext_vector_type(8)))  float    v8f;

__device__ __forceinline__ float signf(float x) {
    return (x > 0.0f) ? 1.0f : ((x < 0.0f) ? -1.0f : 0.0f);
}

// align_corners grid coords on [-1,1] -> [0,3], clamped, linear weights
__device__ __forceinline__ void gridc(float c, int& i0, int& i1, float& w0, float& w1) {
    float t = (c + 1.0f) * 1.5f;
    t = fminf(fmaxf(t, 0.0f), 3.0f);
    float t0 = floorf(t);
    i0 = (int)t0;
    i1 = (i0 + 1 < 3) ? (i0 + 1) : 3;
    float f = t - t0;
    w0 = 1.0f - f;
    w1 = f;
}

// Open3D ball -> cube volume-preserving map (sphere->cylinder->cube)
__device__ __forceinline__ void ball2cube(float x, float y, float z,
                                          float& ox, float& oy, float& oz) {
    const float eps = 1e-12f;
    float sq    = x * x + y * y + z * z;
    float nrm   = sqrtf(fmaxf(sq, eps));
    float sq_xy = x * x + y * y;

    bool  top    = (1.25f * z * z) > sq_xy;
    float s_top  = sqrtf(3.0f * nrm / (nrm + fabsf(z)));
    float s_side = nrm * rsqrtf(fmaxf(sq_xy, eps));

    float x1 = top ? x * s_top : x * s_side;
    float y1 = top ? y * s_top : y * s_side;
    float z1 = top ? signf(z) * nrm : 1.5f * z;
    if (sq < eps) { x1 = x; y1 = y; z1 = z; }

    float sq_xy1  = x1 * x1 + y1 * y1;
    float norm_xy = sqrtf(fmaxf(sq_xy1, eps));
    bool  xbig    = fabsf(y1) <= fabsf(x1);
    float sx      = signf(x1) * norm_xy;
    float sy      = signf(y1) * norm_xy;
    float safe_x  = (fabsf(x1) > eps) ? x1 : 1.0f;
    float safe_y  = (fabsf(y1) > eps) ? y1 : 1.0f;

    const float FOUR_OVER_PI = 1.2732395447351628f;
    float x2 = xbig ? sx : sy * FOUR_OVER_PI * atanf(x1 / safe_y);
    float y2 = xbig ? sx * FOUR_OVER_PI * atanf(y1 / safe_x) : sy;
    if (sq_xy1 < eps) { x2 = x1; y2 = y1; }

    ox = x2; oy = y2; oz = z1;
}

__global__ __launch_bounds__(BLOCK)
void cconv_wmma_kernel(const float* __restrict__ pos,
                       const float* __restrict__ vel,
                       const float* __restrict__ kern,   // [64][3][3] = B[k=g*3+ci][co]
                       const int* __restrict__ nidx,
                       const unsigned char* __restrict__ nmask,
                       float* __restrict__ out)          // [N][3]
{
    __shared__ float Alds[BLOCK * ASTRIDE];   // ~48.25 KB

    const int tid = threadIdx.x;
    const int i   = blockIdx.x * BLOCK + tid;

    float* Arow = &Alds[tid * ASTRIDE];
    #pragma unroll 8
    for (int q = 0; q < KDIM; ++q) Arow[q] = 0.0f;

    // ---------------- Phase 1: neighbor-pair accumulation ----------------
    if (i < N_PART) {
        const float px = pos[i * 3 + 0];
        const float py = pos[i * 3 + 1];
        const float pz = pos[i * 3 + 2];
        const float inv_r  = 1.0f / RADIUS_F;
        const float inv_r2 = inv_r * inv_r;
        const int   base   = i * KMAX;

        for (int k = 0; k < KMAX; ++k) {
            if (!nmask[base + k]) break;   // mask is prefix-true per row
            const int j = nidx[base + k];

            const float rx = pos[j * 3 + 0] - px;
            const float ry = pos[j * 3 + 1] - py;
            const float rz = pos[j * 3 + 2] - pz;
            const float d2 = rx * rx + ry * ry + rz * rz;

            // poly6 window on normalized squared distance
            float t   = 1.0f - d2 * inv_r2;
            float imp = t * t * t;
            imp = fminf(fmaxf(imp, 0.0f), 1.0f);

            float cx, cy, cz;
            ball2cube(rx * inv_r, ry * inv_r, rz * inv_r, cx, cy, cz);

            int   xi0, xi1, yi0, yi1, zi0, zi1;
            float wx0, wx1, wy0, wy1, wz0, wz1;
            gridc(cx, xi0, xi1, wx0, wx1);
            gridc(cy, yi0, yi1, wy0, wy1);
            gridc(cz, zi0, zi1, wz0, wz1);

            const float p0 = vel[j * 3 + 0] * imp;
            const float p1 = vel[j * 3 + 1] * imp;
            const float p2 = vel[j * 3 + 2] * imp;

            #define CORNER(zi_, wz_, yi_, wy_, xi_, wx_)                    \
                do {                                                        \
                    const int   c_ = ((zi_) * 4 + (yi_)) * 4 + (xi_);       \
                    const float w_ = (wz_) * (wy_) * (wx_);                 \
                    float* ap = Arow + c_ * 3;                              \
                    ap[0] += w_ * p0; ap[1] += w_ * p1; ap[2] += w_ * p2;   \
                } while (0)

            CORNER(zi0, wz0, yi0, wy0, xi0, wx0);
            CORNER(zi0, wz0, yi0, wy0, xi1, wx1);
            CORNER(zi0, wz0, yi1, wy1, xi0, wx0);
            CORNER(zi0, wz0, yi1, wy1, xi1, wx1);
            CORNER(zi1, wz1, yi0, wy0, xi0, wx0);
            CORNER(zi1, wz1, yi0, wy0, xi1, wx1);
            CORNER(zi1, wz1, yi1, wy1, xi0, wx0);
            CORNER(zi1, wz1, yi1, wy1, xi1, wx1);
            #undef CORNER
        }
    }

    __syncthreads();

    // ---------------- Phase 2: WMMA contraction  out = A @ W ----------------
    // Per wave: 32 particles = two 16-row M tiles. K = 192 = 6 blocks of 32.
    const int lane   = tid & 31;
    const int waveId = tid >> 5;
    const int n      = lane & 15;   // output column (valid: n < 3)
    const int hi     = lane >> 4;   // lane half-select

    // B fragments (32x16 f16, same K order as A: element e of the v16h =
    // K = hi*8+e for e<8, K = 16+hi*8+(e-8) for e>=8; column = n).
    v16h bfrag[6];
    #pragma unroll
    for (int kk = 0; kk < 6; ++kk) {
        v16h b;
        #pragma unroll
        for (int e = 0; e < 16; ++e) {
            const int kg = kk * 32 + ((e >> 3) << 4) + hi * 8 + (e & 7); // global K index
            const float w = (n < 3) ? kern[kg * 3 + n] : 0.0f; // zero-pad cols 3..15
            b[e] = (_Float16)w;
        }
        bfrag[kk] = b;
    }

    #pragma unroll
    for (int tile = 0; tile < 2; ++tile) {
        // A fragment rows: lanes 0-15 -> M=lane, lanes 16-31 -> M=lane-16 (==n)
        const float* arow = &Alds[(waveId * 32 + tile * 16 + n) * ASTRIDE];

        v8f acc = {};
        #pragma unroll
        for (int kk = 0; kk < 6; ++kk) {
            // 16-bit A 16x32 layout: v0-3 = K hi*8+0..7, v4-7 = K 16+hi*8+0..7
            v16h a;
            const int k0 = kk * 32 + hi * 8;
            #pragma unroll
            for (int e = 0; e < 8; ++e) a[e]     = (_Float16)arow[k0 + e];
            #pragma unroll
            for (int e = 0; e < 8; ++e) a[8 + e] = (_Float16)arow[k0 + 16 + e];

            acc = __builtin_amdgcn_wmma_f32_16x16x32_f16(
                /*neg_a=*/false, a, /*neg_b=*/false, bfrag[kk],
                /*c_mod=*/(short)0, acc, /*reuse_a=*/false, /*reuse_b=*/false);
        }

        // D layout: lane = column N (=n), VGPR v = row M = hi*8 + v
        if (n < 3) {
            const int pbase = blockIdx.x * BLOCK + waveId * 32 + tile * 16 + hi * 8;
            #pragma unroll
            for (int v = 0; v < 8; ++v) {
                const int p = pbase + v;
                if (p < N_PART) out[p * 3 + n] = acc[v];
            }
        }
    }
}

extern "C" void kernel_launch(void* const* d_in, const int* in_sizes, int n_in,
                              void* d_out, int out_size, void* d_ws, size_t ws_size,
                              hipStream_t stream) {
    (void)in_sizes; (void)n_in; (void)out_size; (void)d_ws; (void)ws_size;

    const float*         pos   = (const float*)d_in[0];
    const float*         vel   = (const float*)d_in[1];
    const float*         kern  = (const float*)d_in[2];
    const int*           nidx  = (const int*)d_in[3];
    const unsigned char* nmask = (const unsigned char*)d_in[4];
    float*               out   = (float*)d_out;

    const int nblk = (N_PART + BLOCK - 1) / BLOCK;
    cconv_wmma_kernel<<<nblk, BLOCK, 0, stream>>>(pos, vel, kern, nidx, nmask, out);
}
